// TransformerEncoderLayer_6614249635916
// MI455X (gfx1250) — compile-verified, matrix operand order checked by reading, not run on hardware
//
#include <hip/hip_runtime.h>
#include <hip/hip_bf16.h>

// ---------------------------------------------------------------------------
// Problem constants (from reference)
// ---------------------------------------------------------------------------
constexpr int S  = 2048;
constexpr int Bb = 4;
constexpr int Dd = 1024;
constexpr int Hh = 16;
constexpr int HD = 64;
constexpr int Ff = 4096;
constexpr float SOFTMAX_SCALE = 0.125f;   // 1/sqrt(64)
constexpr float LN_EPS = 1e-5f;

typedef __attribute__((ext_vector_type(16))) __bf16 v16bf;
typedef __attribute__((ext_vector_type(8)))  __bf16 v8bf;
typedef __attribute__((ext_vector_type(8)))  float  v8f;

__device__ inline v8f wmma_bf16(v16bf a, v16bf b, v8f c) {
  return __builtin_amdgcn_wmma_f32_16x16x32_bf16(false, a, false, b, (short)0, c,
                                                 false, false);
}

// Async copy of 16 bytes global -> LDS (GVS addressing), tracked by ASYNCcnt.
__device__ inline void async_copy16(unsigned lds_off, unsigned gl_byte_off,
                                    unsigned long long base) {
  asm volatile("global_load_async_to_lds_b128 %0, %1, %2"
               :: "v"(lds_off), "v"(gl_byte_off), "s"(base)
               : "memory");
}
__device__ inline void wait_async0() {
  asm volatile("s_wait_asynccnt 0x0" ::: "memory");
}
__device__ inline void wait_async4() {
  asm volatile("s_wait_asynccnt 0x4" ::: "memory");
}
__device__ inline unsigned lds_addr(const void* p) {
  return (unsigned)(uintptr_t)p;  // flat LDS address: offset is addr[31:0]
}

// Load a 16x32 bf16 WMMA A/B fragment from a row-major LDS tile.
// Per ISA 7.12.2 (16-bit A layout, wave32): lane l (r = l&15, hi = l>>4) holds
// K runs [k0+hi*8, +8) in VGPR0-3 and [k0+16+hi*8, +8) in VGPR4-7.
__device__ inline v16bf ldfrag(const __bf16* base, int row0, int stride, int k0,
                               int lane) {
  const int r  = row0 + (lane & 15);
  const int hi = (lane >> 4) & 1;
  const __bf16* p = base + r * stride + k0 + hi * 8;
  union { v16bf v; v8bf h[2]; } u;
  u.h[0] = *(const v8bf*)(p);
  u.h[1] = *(const v8bf*)(p + 16);
  return u.v;
}

// ---------------------------------------------------------------------------
// fp32 -> bf16 bulk conversion (one-shot for weights / input state)
// ---------------------------------------------------------------------------
__global__ __launch_bounds__(256) void cvt_bf16_kernel(
    const float* __restrict__ in, __bf16* __restrict__ out, int n) {
  const int i = (blockIdx.x * 256 + threadIdx.x) * 8;
  if (i + 8 <= n) {
    float4 a0 = *(const float4*)(in + i);
    float4 a1 = *(const float4*)(in + i + 4);
    v8bf x;
    x[0] = (__bf16)a0.x; x[1] = (__bf16)a0.y; x[2] = (__bf16)a0.z; x[3] = (__bf16)a0.w;
    x[4] = (__bf16)a1.x; x[5] = (__bf16)a1.y; x[6] = (__bf16)a1.z; x[7] = (__bf16)a1.w;
    *(v8bf*)(out + i) = x;
  }
}

// ---------------------------------------------------------------------------
// GEMM: C[M,N] = act(A[M,K] @ W[N,K]^T + bias[N]); A,W bf16; out fp32 and/or
// bf16. 256 threads / 8 waves, 128x128 tile, K-step 32, double-buffered LDS
// fed by global_load_async_to_lds_b128 (4 chunks of 16B per thread per step).
// ---------------------------------------------------------------------------
__global__ __launch_bounds__(256) void gemm_bf16_kernel(
    const __bf16* __restrict__ A, const __bf16* __restrict__ W,
    const float* __restrict__ bias, float* __restrict__ Cf,
    __bf16* __restrict__ Cb, int M, int N, int K, int relu) {
  constexpr int LDT = 40;                 // bf16 elems per LDS row (80B)
  constexpr int TILE = 128 * LDT;         // elems per tile per buffer
  constexpr unsigned TILE_B = TILE * 2;   // 10240 bytes
  __shared__ __bf16 As[2 * TILE];
  __shared__ __bf16 Bs[2 * TILE];

  const int tid  = threadIdx.x;
  const int lane = tid & 31;
  const int wave = tid >> 5;
  const int wm   = wave & 1;
  const int wn   = wave >> 1;
  const int m0   = blockIdx.y * 128;
  const int n0   = blockIdx.x * 128;

  v8f acc[4][2];
  const v8f vzero = {0.f, 0.f, 0.f, 0.f, 0.f, 0.f, 0.f, 0.f};
#pragma unroll
  for (int i = 0; i < 4; ++i)
#pragma unroll
    for (int j = 0; j < 2; ++j) acc[i][j] = vzero;

  // Async staging: tile = 128 rows x 64B; chunk = 16B. Thread handles rows
  // r0 = tid>>2 and r0+64, sub-chunk (tid&3), for both A and B tiles.
  const int r0  = tid >> 2;
  const int r1  = r0 + 64;
  const unsigned sb = (unsigned)((tid & 3) * 16);
  const unsigned long long abase = (unsigned long long)(uintptr_t)A;
  const unsigned long long wbase = (unsigned long long)(uintptr_t)W;
  const unsigned asl = lds_addr(&As[0]);
  const unsigned bsl = lds_addr(&Bs[0]);

  auto issue_tiles = [&](int buf, int k0) {
    const unsigned la = asl + (unsigned)buf * TILE_B;
    const unsigned lb = bsl + (unsigned)buf * TILE_B;
    const unsigned kb = (unsigned)(k0 * 2);
    async_copy16(la + (unsigned)r0 * 80u + sb,
                 (unsigned)((m0 + r0) * K) * 2u + kb + sb, abase);
    async_copy16(la + (unsigned)r1 * 80u + sb,
                 (unsigned)((m0 + r1) * K) * 2u + kb + sb, abase);
    async_copy16(lb + (unsigned)r0 * 80u + sb,
                 (unsigned)((n0 + r0) * K) * 2u + kb + sb, wbase);
    async_copy16(lb + (unsigned)r1 * 80u + sb,
                 (unsigned)((n0 + r1) * K) * 2u + kb + sb, wbase);
  };

  issue_tiles(0, 0);
  const int nsteps = K / 32;
  for (int s = 0; s < nsteps; ++s) {
    const int buf = s & 1;
    const bool more = (s + 1 < nsteps);
    if (more) issue_tiles(buf ^ 1, (s + 1) * 32);  // next tile in flight
    if (more) wait_async4(); else wait_async0();   // current tile landed
    __syncthreads();

    const __bf16* Ab = &As[buf * TILE];
    const __bf16* Wb = &Bs[buf * TILE];
    v16bf af[4], bfr[2];
#pragma unroll
    for (int i = 0; i < 4; ++i) af[i] = ldfrag(Ab, wm * 64 + 16 * i, LDT, 0, lane);
#pragma unroll
    for (int j = 0; j < 2; ++j) bfr[j] = ldfrag(Wb, wn * 32 + 16 * j, LDT, 0, lane);
#pragma unroll
    for (int i = 0; i < 4; ++i)
#pragma unroll
      for (int j = 0; j < 2; ++j) acc[i][j] = wmma_bf16(af[i], bfr[j], acc[i][j]);
    __syncthreads();  // frag reads done before next async overwrite
  }

  const int ln15  = lane & 15;
  const int mhalf = (lane >> 4) * 8;
#pragma unroll
  for (int i = 0; i < 4; ++i) {
#pragma unroll
    for (int j = 0; j < 2; ++j) {
      const int mm = m0 + wm * 64 + 16 * i + mhalf;
      const int nn = n0 + wn * 32 + 16 * j + ln15;
      const float bv = bias[nn];
#pragma unroll
      for (int e = 0; e < 8; ++e) {
        float v = acc[i][j][e] + bv;
        if (relu) v = fmaxf(v, 0.f);
        if (Cf) Cf[(size_t)(mm + e) * N + nn] = v;
        if (Cb) Cb[(size_t)(mm + e) * N + nn] = (__bf16)v;
      }
    }
  }
}

// ---------------------------------------------------------------------------
// Flash attention over bf16 Q/K/V. Grid: (S/64, B*H). 256 threads / 8 waves.
// Q/K tiles staged with async-to-LDS; V staged transposed via VGPRs.
// ---------------------------------------------------------------------------
__global__ __launch_bounds__(256) void attention_kernel(
    const __bf16* __restrict__ q, const __bf16* __restrict__ k,
    const __bf16* __restrict__ v, const unsigned char* __restrict__ mask,
    __bf16* __restrict__ out) {
  constexpr int LDT = 72;  // bf16 row stride (144B)
  __shared__ __bf16 Qs[64 * LDT];
  __shared__ __bf16 Ks[64 * LDT];
  __shared__ __bf16 Vts[64 * LDT];  // V transposed: row = head dim, col = key
  __shared__ __bf16 Ps[64 * LDT];   // probabilities
  __shared__ float  Ssc[64 * 68];   // raw scores
  __shared__ float  rowm[64], rowl[64], rowfac[64];

  const int tid  = threadIdx.x;
  const int lane = tid & 31;
  const int wave = tid >> 5;
  const int wm_s = wave >> 1;        // 0..3 : 16-row band
  const int wn_s = wave & 1;         // 0..1 : 32-col band
  const int qb   = blockIdx.x;
  const int b    = blockIdx.y >> 4;
  const int h    = blockIdx.y & 15;

  const int r  = tid >> 2;           // 0..63
  const int c4 = (tid & 3) * 16;     // 0,16,32,48

  // Async chunk coords for 64-row x 128B tiles: chunks tid and tid+256.
  const int rq0 = tid >> 3, sq = (tid & 7) * 16;  // rq1 = rq0 + 32
  const unsigned long long qbase = (unsigned long long)(uintptr_t)q;
  const unsigned long long kbase = (unsigned long long)(uintptr_t)k;
  const unsigned qsl = lds_addr(&Qs[0]);
  const unsigned ksl = lds_addr(&Ks[0]);

  // ---- stage Q via async copy
  {
    const unsigned g0 =
        (unsigned)((((qb * 64 + rq0) * Bb + b) * Dd + h * HD) * 2) + sq;
    const unsigned g1 =
        (unsigned)((((qb * 64 + rq0 + 32) * Bb + b) * Dd + h * HD) * 2) + sq;
    async_copy16(qsl + (unsigned)rq0 * 144u + sq, g0, qbase);
    async_copy16(qsl + (unsigned)(rq0 + 32) * 144u + sq, g1, qbase);
  }
  if (tid < 64) { rowm[tid] = -1e30f; rowl[tid] = 0.f; }

  const v8f vzero = {0.f, 0.f, 0.f, 0.f, 0.f, 0.f, 0.f, 0.f};
  v8f oacc[2] = {vzero, vzero};

  const int ln15  = lane & 15;
  const int mhalf = (lane >> 4) * 8;

  for (int t = 0; t < S / 64; ++t) {
    __syncthreads();  // previous iteration's LDS readers done

    // ---- stage K (async) and V (transposed, via VGPRs)
    {
      const unsigned g0 =
          (unsigned)((((t * 64 + rq0) * Bb + b) * Dd + h * HD) * 2) + sq;
      const unsigned g1 =
          (unsigned)((((t * 64 + rq0 + 32) * Bb + b) * Dd + h * HD) * 2) + sq;
      async_copy16(ksl + (unsigned)rq0 * 144u + sq, g0, kbase);
      async_copy16(ksl + (unsigned)(rq0 + 32) * 144u + sq, g1, kbase);

      const __bf16* vp = v + ((size_t)(t * 64 + r) * Bb + b) * Dd + h * HD + c4;
      v8bf vlo = *(const v8bf*)(vp);
      v8bf vhi = *(const v8bf*)(vp + 8);
#pragma unroll
      for (int i = 0; i < 8; ++i) {
        Vts[(c4 + i) * LDT + r] = vlo[i];
        Vts[(c4 + 8 + i) * LDT + r] = vhi[i];
      }
    }
    wait_async0();
    __syncthreads();

    // ---- scores: S = Q @ K^T
#pragma unroll
    for (int nt = 0; nt < 2; ++nt) {
      const int col0 = wn_s * 32 + nt * 16;
      v8f sacc = vzero;
#pragma unroll
      for (int kk = 0; kk < HD; kk += 32) {
        v16bf aq = ldfrag(Qs, wm_s * 16, LDT, kk, lane);
        v16bf bk = ldfrag(Ks, col0, LDT, kk, lane);
        sacc = wmma_bf16(aq, bk, sacc);
      }
#pragma unroll
      for (int e = 0; e < 8; ++e)
        Ssc[(wm_s * 16 + mhalf + e) * 68 + col0 + ln15] = sacc[e];
    }
    __syncthreads();

    // ---- online softmax (4 lanes per row, 16 cols each)
    {
      const float mold = rowm[r];
      float sv[16];
      float mx = -3.0e38f;
#pragma unroll
      for (int i = 0; i < 16; ++i) {
        const int kg = t * 64 + c4 + i;
        const float sc = mask[b * S + kg] ? -3.0e38f
                                          : Ssc[r * 68 + c4 + i] * SOFTMAX_SCALE;
        sv[i] = sc;
        mx = fmaxf(mx, sc);
      }
      mx = fmaxf(mx, __shfl_xor(mx, 1));
      mx = fmaxf(mx, __shfl_xor(mx, 2));
      const float mnew = fmaxf(mold, mx);
      const float fac  = __expf(mold - mnew);
      float ls = 0.f;
#pragma unroll
      for (int i = 0; i < 16; ++i) {
        const float p = __expf(sv[i] - mnew);
        ls += p;
        Ps[r * LDT + c4 + i] = (__bf16)p;
      }
      ls += __shfl_xor(ls, 1);
      ls += __shfl_xor(ls, 2);
      if ((tid & 3) == 0) {
        rowm[r]   = mnew;
        rowfac[r] = fac;
        rowl[r]   = rowl[r] * fac + ls;
      }
    }
    __syncthreads();

    // ---- rescale O, accumulate P @ V
#pragma unroll
    for (int e = 0; e < 8; ++e) {
      const float fac = rowfac[wm_s * 16 + mhalf + e];
      oacc[0][e] *= fac;
      oacc[1][e] *= fac;
    }
#pragma unroll
    for (int nt = 0; nt < 2; ++nt) {
      const int col0 = wn_s * 32 + nt * 16;
#pragma unroll
      for (int kk = 0; kk < 64; kk += 32) {
        v16bf ap = ldfrag(Ps, wm_s * 16, LDT, kk, lane);
        v16bf bv = ldfrag(Vts, col0, LDT, kk, lane);
        oacc[nt] = wmma_bf16(ap, bv, oacc[nt]);
      }
    }
  }
  __syncthreads();

  // ---- finalize: O /= l, write bf16 (feeds Wo GEMM)
#pragma unroll
  for (int nt = 0; nt < 2; ++nt) {
    const int col0 = wn_s * 32 + nt * 16;
#pragma unroll
    for (int e = 0; e < 8; ++e) {
      const int mrow = wm_s * 16 + mhalf + e;
      const float linv = 1.f / rowl[mrow];
      out[((size_t)(qb * 64 + mrow) * Bb + b) * Dd + h * HD + col0 + ln15] =
          (__bf16)(oacc[nt][e] * linv);
    }
  }
}

// ---------------------------------------------------------------------------
// out = LayerNorm(x + res) * g + beta; optional bf16 copy for next GEMM.
// ---------------------------------------------------------------------------
__global__ __launch_bounds__(256) void add_ln_kernel(
    const float* __restrict__ x, const float* __restrict__ res,
    const float* __restrict__ g, const float* __restrict__ beta,
    float* __restrict__ outf, __bf16* __restrict__ outb) {
  const int row = blockIdx.x;
  const float* xp = x + (size_t)row * Dd;
  const float* rp = res + (size_t)row * Dd;
  float vals[4];
  float s = 0.f, ss = 0.f;
#pragma unroll
  for (int i = 0; i < 4; ++i) {
    const int idx = threadIdx.x + i * 256;
    const float v = xp[idx] + rp[idx];
    vals[i] = v;
    s += v;
    ss += v * v;
  }
#pragma unroll
  for (int m = 1; m <= 16; m <<= 1) {
    s += __shfl_xor(s, m);
    ss += __shfl_xor(ss, m);
  }
  __shared__ float sbuf[8], ssbuf[8];
  const int wave = threadIdx.x >> 5;
  const int lane = threadIdx.x & 31;
  if (lane == 0) { sbuf[wave] = s; ssbuf[wave] = ss; }
  __syncthreads();
  if (wave == 0) {
    float a  = (lane < 8) ? sbuf[lane] : 0.f;
    float bb = (lane < 8) ? ssbuf[lane] : 0.f;
#pragma unroll
    for (int m = 1; m <= 4; m <<= 1) {
      a += __shfl_xor(a, m);
      bb += __shfl_xor(bb, m);
    }
    if (lane == 0) { sbuf[0] = a; ssbuf[0] = bb; }
  }
  __syncthreads();
  const float mean = sbuf[0] * (1.f / Dd);
  const float var  = ssbuf[0] * (1.f / Dd) - mean * mean;
  const float rstd = rsqrtf(var + LN_EPS);
#pragma unroll
  for (int i = 0; i < 4; ++i) {
    const int idx = threadIdx.x + i * 256;
    const float o = (vals[i] - mean) * rstd * g[idx] + beta[idx];
    outf[(size_t)row * Dd + idx] = o;
    if (outb) outb[(size_t)row * Dd + idx] = (__bf16)o;
  }
}

// ---------------------------------------------------------------------------
// Host-side orchestration (graph-capture safe: launches only).
// ---------------------------------------------------------------------------
extern "C" void kernel_launch(void* const* d_in, const int* in_sizes, int n_in,
                              void* d_out, int out_size, void* d_ws, size_t ws_size,
                              hipStream_t stream) {
  (void)in_sizes; (void)n_in; (void)out_size; (void)ws_size;
  const float*         state = (const float*)d_in[0];
  const unsigned char* mask  = (const unsigned char*)d_in[1];
  const float* Wq = (const float*)d_in[2];
  const float* bq = (const float*)d_in[3];
  const float* Wk = (const float*)d_in[4];
  const float* bk = (const float*)d_in[5];
  const float* Wv = (const float*)d_in[6];
  const float* bv = (const float*)d_in[7];
  const float* Wo = (const float*)d_in[8];
  const float* bo = (const float*)d_in[9];
  const float* ln1_g = (const float*)d_in[10];
  const float* ln1_b = (const float*)d_in[11];
  const float* W1 = (const float*)d_in[12];
  const float* b1 = (const float*)d_in[13];
  const float* W2 = (const float*)d_in[14];
  const float* b2 = (const float*)d_in[15];
  const float* ln2_g = (const float*)d_in[16];
  const float* ln2_b = (const float*)d_in[17];

  char* ws = (char*)d_ws;
  const size_t MBu = 1ull << 20;
  __bf16* stateb  = (__bf16*)(ws + 0 * MBu);     // 16 MB
  __bf16* Wqb     = (__bf16*)(ws + 16 * MBu);    // 2 MB
  __bf16* Wkb     = (__bf16*)(ws + 18 * MBu);    // 2 MB
  __bf16* Wvb     = (__bf16*)(ws + 20 * MBu);    // 2 MB
  __bf16* Wob     = (__bf16*)(ws + 22 * MBu);    // 2 MB
  __bf16* W1b     = (__bf16*)(ws + 24 * MBu);    // 8 MB
  __bf16* W2b     = (__bf16*)(ws + 32 * MBu);    // 8 MB
  __bf16* qb      = (__bf16*)(ws + 40 * MBu);    // 16 MB
  __bf16* kb      = (__bf16*)(ws + 56 * MBu);    // 16 MB
  __bf16* vb      = (__bf16*)(ws + 72 * MBu);    // 16 MB
  __bf16* attnb   = (__bf16*)(ws + 88 * MBu);    // 16 MB
  float*  proj    = (float*)(ws + 104 * MBu);    // 32 MB
  float*  state1f = (float*)(ws + 136 * MBu);    // 32 MB
  __bf16* state1b = (__bf16*)(ws + 168 * MBu);   // 16 MB
  __bf16* ffhb    = (__bf16*)(ws + 184 * MBu);   // 64 MB
  float*  ffo     = (float*)(ws + 248 * MBu);    // 32 MB

  const int M = S * Bb;  // 8192 ([S,B,D] flattens to [S*B, D])
  dim3 blk(256);
  dim3 gD(Dd / 128, M / 128);
  dim3 gF(Ff / 128, M / 128);

  // One-shot bf16 conversions (weights + input state)
  cvt_bf16_kernel<<<dim3((M * Dd) / 2048), blk, 0, stream>>>(state, stateb, M * Dd);
  cvt_bf16_kernel<<<dim3((Dd * Dd) / 2048), blk, 0, stream>>>(Wq, Wqb, Dd * Dd);
  cvt_bf16_kernel<<<dim3((Dd * Dd) / 2048), blk, 0, stream>>>(Wk, Wkb, Dd * Dd);
  cvt_bf16_kernel<<<dim3((Dd * Dd) / 2048), blk, 0, stream>>>(Wv, Wvb, Dd * Dd);
  cvt_bf16_kernel<<<dim3((Dd * Dd) / 2048), blk, 0, stream>>>(Wo, Wob, Dd * Dd);
  cvt_bf16_kernel<<<dim3((Ff * Dd) / 2048), blk, 0, stream>>>(W1, W1b, Ff * Dd);
  cvt_bf16_kernel<<<dim3((Dd * Ff) / 2048), blk, 0, stream>>>(W2, W2b, Dd * Ff);

  // QKV projections (bf16 out, consumed by attention)
  gemm_bf16_kernel<<<gD, blk, 0, stream>>>(stateb, Wqb, bq, nullptr, qb, M, Dd, Dd, 0);
  gemm_bf16_kernel<<<gD, blk, 0, stream>>>(stateb, Wkb, bk, nullptr, kb, M, Dd, Dd, 0);
  gemm_bf16_kernel<<<gD, blk, 0, stream>>>(stateb, Wvb, bv, nullptr, vb, M, Dd, Dd, 0);
  // Attention (bf16 out, consumed by Wo GEMM)
  attention_kernel<<<dim3(S / 64, Bb * Hh), blk, 0, stream>>>(qb, kb, vb, mask, attnb);
  // Output projection (fp32 out, feeds LN1) + residual LN -> state1 (f32+bf16)
  gemm_bf16_kernel<<<gD, blk, 0, stream>>>(attnb, Wob, bo, proj, nullptr, M, Dd, Dd, 0);
  add_ln_kernel<<<dim3(M), blk, 0, stream>>>(proj, state, ln1_g, ln1_b, state1f,
                                             state1b);
  // FFN: relu GEMM (bf16 out) -> GEMM (fp32 out) -> residual LN -> d_out
  gemm_bf16_kernel<<<gF, blk, 0, stream>>>(state1b, W1b, b1, nullptr, ffhb,
                                           M, Ff, Dd, 1);
  gemm_bf16_kernel<<<gD, blk, 0, stream>>>(ffhb, W2b, b2, ffo, nullptr,
                                           M, Dd, Ff, 0);
  add_ln_kernel<<<dim3(M), blk, 0, stream>>>(ffo, state1f, ln2_g, ln2_b,
                                             (float*)d_out, nullptr);
}
